// Quantizer_19731079757832
// MI455X (gfx1250) — matrix operand order repaired and checked by reading, not compiled, not run on hardware
//
#include <hip/hip_runtime.h>
#include <hip/hip_bf16.h>

typedef __attribute__((ext_vector_type(16))) _Float16 v16h;
typedef __attribute__((ext_vector_type(8)))  float    v8f;

// Problem constants (from reference): x [16,256,64,64], weight [2048,256]
#define N_ROWS   65536      // 16*64*64 flattened vectors
#define DIM      256
#define N_CODE   2048
#define N_ELEM   16777216   // 16*256*64*64
#define HW       4096       // 64*64
#define CODE_TILES (N_CODE/16)   // 128

// ---------------------------------------------------------------------------
// Kernel 0: convert codebook to f16 + per-code squared norm. One block = one code.
// ---------------------------------------------------------------------------
__global__ __launch_bounds__(256)
void qz_prep(const float* __restrict__ w, _Float16* __restrict__ wh,
             float* __restrict__ wnorm) {
    __shared__ float red[256];
    const int code = blockIdx.x;
    const int t    = threadIdx.x;
    float v = w[(size_t)code * DIM + t];
    wh[(size_t)code * DIM + t] = (_Float16)v;
    red[t] = v * v;
    __syncthreads();
    #pragma unroll
    for (int off = 128; off > 0; off >>= 1) {
        if (t < off) red[t] += red[t + off];
        __syncthreads();
    }
    if (t == 0) wnorm[code] = red[0];
}

// ---------------------------------------------------------------------------
// Kernel 1: WMMA distance GEMM + per-row argmin.
// Block = 256 threads = 8 waves; each wave owns 32 rows (two 16-row M-tiles,
// register-blocked) so every B fragment read from LDS feeds TWO WMMAs.
// All waves share the 16-code B tile staged in LDS.
// ---------------------------------------------------------------------------
__global__ __launch_bounds__(256)
void qz_argmin(const float* __restrict__ x, const _Float16* __restrict__ wh,
               const float* __restrict__ wnorm, int* __restrict__ idx_out) {
    __shared__ __align__(32) _Float16 Btile[16 * DIM];   // 8 KB

    const int lane = threadIdx.x & 31;
    const int col  = lane & 15;          // A-row within tile / B-col (code)
    const int kh   = lane >> 4;          // K-half select (contiguous k order used here)
    const int wave = threadIdx.x >> 5;
    const int row_base = (blockIdx.x * 8 + wave) * 32;

    // ---- Load A fragments (this lane's rows, all K) once; reused 128x ----
    // flat[n, c] = x[b*2^20 + c*4096 + hw],  n = b*4096 + hw
    const int n0 = row_base + col;
    const int n1 = row_base + 16 + col;
    const float* xb0 = x + ((size_t)(n0 >> 12) << 20) + (size_t)(n0 & (HW - 1));
    const float* xb1 = x + ((size_t)(n1 >> 12) << 20) + (size_t)(n1 & (HW - 1));
    v16h afrag0[8], afrag1[8];
    #pragma unroll
    for (int kk = 0; kk < 8; ++kk) {
        v16h a0, a1;
        #pragma unroll
        for (int j = 0; j < 8; ++j) {
            // contiguous k order (code's choice, not ISA table): VGPR j holds K pair; kh selects K-half
            const int k0 = kk * 32 + kh * 16 + 2 * j;
            a0[2 * j]     = (_Float16)xb0[(size_t)k0 * HW];
            a0[2 * j + 1] = (_Float16)xb0[(size_t)(k0 + 1) * HW];
            a1[2 * j]     = (_Float16)xb1[(size_t)k0 * HW];
            a1[2 * j + 1] = (_Float16)xb1[(size_t)(k0 + 1) * HW];
        }
        afrag0[kk] = a0;
        afrag1[kk] = a1;
    }

    // Per-lane running best over the codes this column position sees.
    float bestv0[8], bestv1[8];
    int   besti0[8], besti1[8];
    #pragma unroll
    for (int j = 0; j < 8; ++j) {
        bestv0[j] = 3.4e38f; besti0[j] = 0;
        bestv1[j] = 3.4e38f; besti1[j] = 0;
    }

    for (int t = 0; t < CODE_TILES; ++t) {
        __syncthreads();   // previous tile fully consumed
        // Cooperative stage of 16 codes x 256 ch (f16): 256 thr x 32B
        *(v16h*)&Btile[threadIdx.x * 16] =
            *(const v16h*)&wh[(size_t)t * 16 * DIM + threadIdx.x * 16];
        if (t + 1 < CODE_TILES)
            __builtin_prefetch(&wh[(size_t)(t + 1) * 16 * DIM + threadIdx.x * 16], 0, 3);
        __syncthreads();   // tile visible

        const float wn = wnorm[t * 16 + col];

        v8f acc0 = {}, acc1 = {};
        #pragma unroll
        for (int kk = 0; kk < 8; ++kk) {
            // B uses same contiguous k order as A: col = lane&15, kh selects K 0-15 / 16-31,
            // VGPR j holds K pair (contiguous) -> 32B-aligned v16h LDS read.
            // One B fragment feeds two independent WMMA accumulation chains.
            const v16h b = *(const v16h*)&Btile[col * DIM + kk * 32 + kh * 16];
            acc0 = __builtin_amdgcn_wmma_f32_16x16x32_f16(
                false, afrag0[kk], false, b, (short)0, acc0, false, false);
            acc1 = __builtin_amdgcn_wmma_f32_16x16x32_f16(
                false, afrag1[kk], false, b, (short)0, acc1, false, false);
        }

        const int code = t * 16 + col;
        #pragma unroll
        for (int j = 0; j < 8; ++j) {
            const float d0 = wn - 2.0f * acc0[j];  // + ||x||^2 is argmin-invariant
            const float d1 = wn - 2.0f * acc1[j];
            if (d0 < bestv0[j]) { bestv0[j] = d0; besti0[j] = code; }
            if (d1 < bestv1[j]) { bestv1[j] = d1; besti1[j] = code; }
        }
    }

    // Reduce across the 16 columns (within each half-wave); tie -> smaller idx
    #pragma unroll
    for (int j = 0; j < 8; ++j) {
        #pragma unroll
        for (int m = 8; m >= 1; m >>= 1) {
            const float ov0 = __shfl_xor(bestv0[j], m, 32);
            const int   oi0 = __shfl_xor(besti0[j], m, 32);
            if (ov0 < bestv0[j] || (ov0 == bestv0[j] && oi0 < besti0[j])) {
                bestv0[j] = ov0; besti0[j] = oi0;
            }
            const float ov1 = __shfl_xor(bestv1[j], m, 32);
            const int   oi1 = __shfl_xor(besti1[j], m, 32);
            if (ov1 < bestv1[j] || (ov1 == bestv1[j] && oi1 < besti1[j])) {
                bestv1[j] = ov1; besti1[j] = oi1;
            }
        }
        // C/D layout: reg j, lanes 0-15 -> row j; lanes 16-31 -> row j+8
        if (col == 0) {
            idx_out[row_base + kh * 8 + j]      = besti0[j];
            idx_out[row_base + 16 + kh * 8 + j] = besti1[j];
        }
    }
}

// ---------------------------------------------------------------------------
// Kernel 2: gather quantized output + fused fp32 loss partial sums.
// ---------------------------------------------------------------------------
__global__ __launch_bounds__(256)
void qz_gather(const float* __restrict__ x, const float* __restrict__ w,
               const int* __restrict__ idx, float* __restrict__ out,
               float* __restrict__ partials) {
    __shared__ float red[256];
    const size_t base = (size_t)blockIdx.x * 2048 + threadIdx.x;
    float s = 0.0f;
    #pragma unroll
    for (int i = 0; i < 8; ++i) {
        const size_t e  = base + (size_t)i * 256;          // [b, c, h, w] flat
        const size_t b  = e >> 20;
        const size_t c  = (e >> 12) & 255;
        const size_t hw = e & (HW - 1);
        const int    id = idx[(b << 12) + hw];
        const float  q  = w[(size_t)id * DIM + c];
        const float  xv = x[e];
        out[e] = q;                                        // == quantized_st
        const float d = q - xv;
        s += d * d;
    }
    red[threadIdx.x] = s;
    __syncthreads();
    #pragma unroll
    for (int off = 128; off > 0; off >>= 1) {
        if (threadIdx.x < off) red[threadIdx.x] += red[threadIdx.x + off];
        __syncthreads();
    }
    if (threadIdx.x == 0) partials[blockIdx.x] = red[0];
}

// ---------------------------------------------------------------------------
// Kernel 3: deterministic final reduction of 8192 partials -> latent loss.
// ---------------------------------------------------------------------------
__global__ __launch_bounds__(256)
void qz_finalize(const float* __restrict__ partials, float* __restrict__ loss) {
    __shared__ float red[256];
    float s = 0.0f;
    for (int i = threadIdx.x; i < 8192; i += 256) s += partials[i];
    red[threadIdx.x] = s;
    __syncthreads();
    #pragma unroll
    for (int off = 128; off > 0; off >>= 1) {
        if (threadIdx.x < off) red[threadIdx.x] += red[threadIdx.x + off];
        __syncthreads();
    }
    if (threadIdx.x == 0) loss[0] = red[0] * (0.25f / (float)N_ELEM);
}

// ---------------------------------------------------------------------------
extern "C" void kernel_launch(void* const* d_in, const int* in_sizes, int n_in,
                              void* d_out, int out_size, void* d_ws, size_t ws_size,
                              hipStream_t stream) {
    const float* x = (const float*)d_in[0];                 // [16,256,64,64]
    const float* w = (const float*)d_in[1];                 // [2048,256]
    float* out = (float*)d_out;                             // 16777216 + 1 (loss)

    // Workspace layout
    char* ws = (char*)d_ws;
    _Float16* wh    = (_Float16*)(ws);                      // 1,048,576 B
    float*    wnorm = (float*)(ws + 1048576);               //     8,192 B
    int*      idx   = (int*)(ws + 1056768);                 //   262,144 B
    float*    part  = (float*)(ws + 1318912);               //    32,768 B

    qz_prep   <<<N_CODE,            256, 0, stream>>>(w, wh, wnorm);
    qz_argmin <<<N_ROWS / (8 * 32), 256, 0, stream>>>(x, wh, wnorm, idx);
    qz_gather <<<N_ELEM / 2048,     256, 0, stream>>>(x, w, idx, out, part);
    qz_finalize<<<1,                256, 0, stream>>>(part, out + N_ELEM);
}
